// Edge_node_50869592655525
// MI455X (gfx1250) — matrix operand order repaired and checked by reading, not compiled, not run on hardware
//
#include <hip/hip_runtime.h>
#include <hip/hip_bf16.h>

// ---------------------------------------------------------------------------
// CDNA5 (gfx1250) wave32 WMMA pipeline for the ptensor edge/node block.
// GEMMs run on v_wmma_f32_16x16x32_bf16 (bf16 A/B, f32 accumulate).
// Large intermediates stored bf16 in workspace (memory-bound problem).
// sched_group_barrier pipelines each k-step: all ds_load_b128 fragment loads
// issue as one group, then the WMMA burst, so LDS latency overlaps compute.
// ---------------------------------------------------------------------------

typedef __attribute__((ext_vector_type(16))) __bf16 v16bf;
typedef __attribute__((ext_vector_type(8)))  __bf16 v8bf;
typedef __attribute__((ext_vector_type(8)))  float  v8f;

#define EPSF 1e-5f

#if defined(__has_builtin)
#if __has_builtin(__builtin_amdgcn_sched_group_barrier)
#define SCHED_GROUP(mask, size, id) __builtin_amdgcn_sched_group_barrier(mask, size, id)
#endif
#endif
#ifndef SCHED_GROUP
#define SCHED_GROUP(mask, size, id)
#endif
#define SG_DSREAD 0x100
#define SG_WMMA   0x008

// fragment built from two contiguous 16B LDS chunks (matches CDNA5 16-bit
// A layout: lanes<16 hold K[k0..k0+7],K[k0+16..23]; lanes>=16 the other half;
// B layout: lanes<16 hold K[k0..k0+7],K[k0+16..23], lanes>=16 the rest, N = lane&15)
__device__ __forceinline__ v16bf ld_frag(const __bf16* p0, const __bf16* p1) {
  v8bf a = *(const v8bf*)p0;
  v8bf b = *(const v8bf*)p1;
  v16bf r;
#pragma unroll
  for (int i = 0; i < 8; ++i) { r[i] = a[i]; r[i + 8] = b[i]; }
  return r;
}

__device__ __forceinline__ v8f wmma_bf16(v16bf a, v16bf b, v8f c) {
  return __builtin_amdgcn_wmma_f32_16x16x32_bf16(
      false, a, false, b, (short)0, c, false, false);
}

// ---------------------------------------------------------------------------
// Kernel: edge GEMM1.  A = [edge_rep | gather | gather+partner] [64 x 192]
// built on the fly in LDS (bf16); W = eW1 [192,128] staged transposed in two
// K-halves (LDS budget).  Output X1 (bf16) + per-column sum/sumsq atomics.
// ---------------------------------------------------------------------------
#define AS1 200   // A row stride (bf16 elems): 192 + 8 pad -> 100 dwords, conflict-free
#define WS1 104   // W row stride for 96-wide K chunk: 96 + 8 pad

__global__ __launch_bounds__(128) void k_edge_gemm1(
    const float* __restrict__ edge_rep, const float* __restrict__ node_rep,
    const int* __restrict__ edge_nodes, const float* __restrict__ W,
    __bf16* __restrict__ Xout, float* __restrict__ gsum,
    float* __restrict__ gsq, int rows) {
  __shared__ __bf16 Atile[64 * AS1];     // 25.6 KB
  __shared__ __bf16 Wtile[128 * WS1];    // 26.6 KB
  __shared__ float lsum[128], lsq[128];

  const int tid = threadIdx.x, lane = tid & 31, wave = tid >> 5;
  const int rowBase = blockIdx.x * 64;
  lsum[tid] = 0.f; lsq[tid] = 0.f;

  // ---- build A tile (gather + linmaps), f32 -> bf16 ----
  for (int idx = tid; idx < 64 * 64; idx += 128) {
    int r = idx >> 6, c = idx & 63;
    int g = rowBase + r;
    float er = 0.f, a = 0.f, s = 0.f;
    if (g < rows) {
      int nid = edge_nodes[g];
      int pid = edge_nodes[g ^ 1];           // partner row in the edge pair
      er = edge_rep[(size_t)g * 64 + c];
      a  = node_rep[(size_t)nid * 64 + c];
      s  = a + node_rep[(size_t)pid * 64 + c];
    }
    __bf16* arow = &Atile[r * AS1];
    arow[c]       = (__bf16)er;
    arow[64 + c]  = (__bf16)a;
    arow[128 + c] = (__bf16)s;
  }

  v8f acc[8];
#pragma unroll
  for (int t = 0; t < 8; ++t) acc[t] = v8f{0.f,0.f,0.f,0.f,0.f,0.f,0.f,0.f};

  const int nlane = lane & 15;
  const int arow  = (wave * 16 + nlane) * AS1;
  const int kh  = (lane < 16) ? 0 : 8;    // A-fragment half select
  const int kbh = (lane < 16) ? 0 : 8;    // B-fragment half select

  for (int kc = 0; kc < 192; kc += 96) {
    __syncthreads();
    // stage W chunk transposed: Wtile[n][kk] = W[kc+kk][n]
    for (int idx = tid; idx < 96 * 128; idx += 128) {
      int n = idx & 127, kk = idx >> 7;
      Wtile[n * WS1 + kk] = (__bf16)W[(size_t)(kc + kk) * 128 + n];
    }
    __syncthreads();
#pragma unroll
    for (int kk = 0; kk < 96; kk += 32) {
      // batch-issue all LDS fragment loads, then the WMMA burst
      v16bf af = ld_frag(&Atile[arow + kc + kk + kh],
                         &Atile[arow + kc + kk + 16 + kh]);
      v16bf bfrag[8];
#pragma unroll
      for (int t = 0; t < 8; ++t) {
        const __bf16* wr = &Wtile[(t * 16 + nlane) * WS1 + kk + kbh];
        bfrag[t] = ld_frag(wr, wr + 16);
      }
#pragma unroll
      for (int t = 0; t < 8; ++t) acc[t] = wmma_bf16(af, bfrag[t], acc[t]);
      SCHED_GROUP(SG_DSREAD, 18, 0);   // 2 A + 16 B ds_load_b128 first
      SCHED_GROUP(SG_WMMA,    8, 0);   // then the 8 WMMAs
    }
  }

  // ---- epilogue: store bf16 + column stats ----
  const int roff = (lane < 16) ? 0 : 8;
#pragma unroll
  for (int t = 0; t < 8; ++t) {
    int n = t * 16 + nlane;
    float s = 0.f, q = 0.f;
#pragma unroll
    for (int v = 0; v < 8; ++v) {
      int grow = rowBase + wave * 16 + roff + v;
      if (grow < rows) {
        float d = acc[t][v];
        Xout[(size_t)grow * 128 + n] = (__bf16)d;
        s += d; q += d * d;
      }
    }
    atomicAdd(&lsum[n], s);
    atomicAdd(&lsq[n], q);
  }
  __syncthreads();
  atomicAdd(&gsum[tid], lsum[tid]);
  atomicAdd(&gsq[tid],  lsq[tid]);
}

// ---------------------------------------------------------------------------
// Kernel: node GEMM1.  A = [node_rep | e2n] [64 x 128]; W = nW1 [128,128].
// ---------------------------------------------------------------------------
#define AS2 136   // 128 + 8 pad -> 68 dwords, conflict-free

__global__ __launch_bounds__(128) void k_node_gemm1(
    const float* __restrict__ node_rep, const float* __restrict__ e2n,
    const float* __restrict__ W, __bf16* __restrict__ Xout,
    float* __restrict__ gsum, float* __restrict__ gsq, int rows) {
  __shared__ __bf16 Atile[64 * AS2];     // 17.4 KB
  __shared__ __bf16 Wtile[128 * AS2];    // 34.8 KB
  __shared__ float lsum[128], lsq[128];

  const int tid = threadIdx.x, lane = tid & 31, wave = tid >> 5;
  const int rowBase = blockIdx.x * 64;
  lsum[tid] = 0.f; lsq[tid] = 0.f;

  for (int idx = tid; idx < 64 * 64; idx += 128) {
    int r = idx >> 6, c = idx & 63;
    int g = rowBase + r;
    float a0 = 0.f, a1 = 0.f;
    if (g < rows) {
      a0 = node_rep[(size_t)g * 64 + c];
      a1 = e2n[(size_t)g * 64 + c];
    }
    __bf16* arow = &Atile[r * AS2];
    arow[c]      = (__bf16)a0;
    arow[64 + c] = (__bf16)a1;
  }
  for (int idx = tid; idx < 128 * 128; idx += 128) {
    int n = idx & 127, k = idx >> 7;
    Wtile[n * AS2 + k] = (__bf16)W[(size_t)k * 128 + n];
  }
  __syncthreads();

  v8f acc[8];
#pragma unroll
  for (int t = 0; t < 8; ++t) acc[t] = v8f{0.f,0.f,0.f,0.f,0.f,0.f,0.f,0.f};

  const int nlane = lane & 15;
  const int arow  = (wave * 16 + nlane) * AS2;
  const int kh  = (lane < 16) ? 0 : 8;
  const int kbh = (lane < 16) ? 0 : 8;

#pragma unroll
  for (int kk = 0; kk < 128; kk += 32) {
    v16bf af = ld_frag(&Atile[arow + kk + kh], &Atile[arow + kk + 16 + kh]);
    v16bf bfrag[8];
#pragma unroll
    for (int t = 0; t < 8; ++t) {
      const __bf16* wr = &Wtile[(t * 16 + nlane) * AS2 + kk + kbh];
      bfrag[t] = ld_frag(wr, wr + 16);
    }
#pragma unroll
    for (int t = 0; t < 8; ++t) acc[t] = wmma_bf16(af, bfrag[t], acc[t]);
    SCHED_GROUP(SG_DSREAD, 18, 0);
    SCHED_GROUP(SG_WMMA,    8, 0);
  }

  const int roff = (lane < 16) ? 0 : 8;
#pragma unroll
  for (int t = 0; t < 8; ++t) {
    int n = t * 16 + nlane;
    float s = 0.f, q = 0.f;
#pragma unroll
    for (int v = 0; v < 8; ++v) {
      int grow = rowBase + wave * 16 + roff + v;
      if (grow < rows) {
        float d = acc[t][v];
        Xout[(size_t)grow * 128 + n] = (__bf16)d;
        s += d; q += d * d;
      }
    }
    atomicAdd(&lsum[n], s);
    atomicAdd(&lsq[n], q);
  }
  __syncthreads();
  atomicAdd(&gsum[tid], lsum[tid]);
  atomicAdd(&gsq[tid],  lsq[tid]);
}

// ---------------------------------------------------------------------------
// Kernel: second-layer GEMM (shared by edge & node paths).
// A = relu(BN(Xin)) [64 x 128] (bf16 in LDS); W [128,64] transposed in LDS.
// ---------------------------------------------------------------------------
__global__ __launch_bounds__(128) void k_gemm2(
    const __bf16* __restrict__ Xin, const float* __restrict__ scale,
    const float* __restrict__ shift, const float* __restrict__ W,
    __bf16* __restrict__ Xout, float* __restrict__ gsum,
    float* __restrict__ gsq, int rows) {
  __shared__ __bf16 Atile[64 * AS2];   // 17.4 KB
  __shared__ __bf16 Wtile[64 * AS2];   // 17.4 KB
  __shared__ float lsum[64], lsq[64];

  const int tid = threadIdx.x, lane = tid & 31, wave = tid >> 5;
  const int rowBase = blockIdx.x * 64;
  if (tid < 64) { lsum[tid] = 0.f; lsq[tid] = 0.f; }

  for (int idx = tid; idx < 64 * 128; idx += 128) {
    int r = idx >> 7, c = idx & 127;
    int g = rowBase + r;
    float x = 0.f;
    if (g < rows) {
      x = (float)Xin[(size_t)g * 128 + c];
      x = fmaxf(x * scale[c] + shift[c], 0.f);   // BN + ReLU
    }
    Atile[r * AS2 + c] = (__bf16)x;
  }
  for (int idx = tid; idx < 128 * 64; idx += 128) {
    int n = idx & 63, k = idx >> 6;
    Wtile[n * AS2 + k] = (__bf16)W[(size_t)k * 64 + n];
  }
  __syncthreads();

  v8f acc[4];
#pragma unroll
  for (int t = 0; t < 4; ++t) acc[t] = v8f{0.f,0.f,0.f,0.f,0.f,0.f,0.f,0.f};

  const int nlane = lane & 15;
  const int arow  = (wave * 16 + nlane) * AS2;
  const int kh  = (lane < 16) ? 0 : 8;
  const int kbh = (lane < 16) ? 0 : 8;

#pragma unroll
  for (int kk = 0; kk < 128; kk += 32) {
    v16bf af = ld_frag(&Atile[arow + kk + kh], &Atile[arow + kk + 16 + kh]);
    v16bf bfrag[4];
#pragma unroll
    for (int t = 0; t < 4; ++t) {
      const __bf16* wr = &Wtile[(t * 16 + nlane) * AS2 + kk + kbh];
      bfrag[t] = ld_frag(wr, wr + 16);
    }
#pragma unroll
    for (int t = 0; t < 4; ++t) acc[t] = wmma_bf16(af, bfrag[t], acc[t]);
    SCHED_GROUP(SG_DSREAD, 10, 0);   // 2 A + 8 B loads first
    SCHED_GROUP(SG_WMMA,    4, 0);   // then the 4 WMMAs
  }

  const int roff = (lane < 16) ? 0 : 8;
#pragma unroll
  for (int t = 0; t < 4; ++t) {
    int n = t * 16 + nlane;
    float s = 0.f, q = 0.f;
#pragma unroll
    for (int v = 0; v < 8; ++v) {
      int grow = rowBase + wave * 16 + roff + v;
      if (grow < rows) {
        float d = acc[t][v];
        Xout[(size_t)grow * 64 + n] = (__bf16)d;
        s += d; q += d * d;
      }
    }
    atomicAdd(&lsum[n], s);
    atomicAdd(&lsq[n], q);
  }
  __syncthreads();
  if (tid < 64) {
    atomicAdd(&gsum[tid], lsum[tid]);
    atomicAdd(&gsq[tid],  lsq[tid]);
  }
}

// ---------------------------------------------------------------------------
// Small kernels: stat finalize, zero-fill, epilogues
// ---------------------------------------------------------------------------
__global__ void k_finalize(float* __restrict__ stats, int slot, int C,
                           float inv_cnt, const float* __restrict__ gamma,
                           const float* __restrict__ beta) {
  int i = threadIdx.x;
  if (i >= C) return;
  float m   = stats[slot * 128 + i] * inv_cnt;
  float var = stats[(slot + 1) * 128 + i] * inv_cnt - m * m;
  float sc  = rsqrtf(var + EPSF) * gamma[i];
  stats[(slot + 2) * 128 + i] = sc;
  stats[(slot + 3) * 128 + i] = beta[i] - m * sc;
}

__global__ void k_zero(float* __restrict__ p, long long n) {
  long long i = blockIdx.x * (long long)blockDim.x + threadIdx.x;
  long long s = gridDim.x * (long long)blockDim.x;
  for (; i < n; i += s) p[i] = 0.f;
}

// edge output: BN+ReLU, write edge_out (f32), scatter-add into e2n (L2-resident)
__global__ void k_edge_out(const __bf16* __restrict__ X2,
                           const float* __restrict__ scale,
                           const float* __restrict__ shift,
                           const int* __restrict__ edge_nodes,
                           float* __restrict__ edge_out,
                           float* __restrict__ e2n, long long total) {
  long long i = blockIdx.x * (long long)blockDim.x + threadIdx.x;
  long long s = gridDim.x * (long long)blockDim.x;
  for (; i < total; i += s) {
    int c = (int)(i & 63);
    long long g = i >> 6;
    float y = fmaxf((float)X2[i] * scale[c] + shift[c], 0.f);
    edge_out[i] = y;
    atomicAdd(&e2n[(size_t)edge_nodes[g] * 64 + c], y);
  }
}

__global__ void k_node_out(const __bf16* __restrict__ Y2,
                           const float* __restrict__ scale,
                           const float* __restrict__ shift,
                           float* __restrict__ out, long long total) {
  long long i = blockIdx.x * (long long)blockDim.x + threadIdx.x;
  long long s = gridDim.x * (long long)blockDim.x;
  for (; i < total; i += s) {
    int c = (int)(i & 63);
    out[i] = fmaxf((float)Y2[i] * scale[c] + shift[c], 0.f);
  }
}

// ---------------------------------------------------------------------------
extern "C" void kernel_launch(void* const* d_in, const int* in_sizes, int n_in,
                              void* d_out, int out_size, void* d_ws,
                              size_t ws_size, hipStream_t stream) {
  const float* node_rep  = (const float*)d_in[0];
  const float* edge_rep  = (const float*)d_in[1];
  const int*   edge_nodes= (const int*)  d_in[2];
  const float* eW1 = (const float*)d_in[3];
  const float* eg1 = (const float*)d_in[4];
  const float* eb1 = (const float*)d_in[5];
  const float* eW2 = (const float*)d_in[6];
  const float* eg2 = (const float*)d_in[7];
  const float* eb2 = (const float*)d_in[8];
  const float* nW1 = (const float*)d_in[9];
  const float* ng1 = (const float*)d_in[10];
  const float* nb1 = (const float*)d_in[11];
  const float* nW2 = (const float*)d_in[12];
  const float* ng2 = (const float*)d_in[13];
  const float* nb2 = (const float*)d_in[14];

  const int Nn    = in_sizes[0] / 64;   // nodes
  const int rowsE = in_sizes[2];        // 2E edge rows

  // workspace layout: stats (16 x 128 f32) | e2n (N x 64 f32) | bf16 buffers
  float* stats = (float*)d_ws;
  float* e2n   = stats + 16 * 128;
  __bf16* X1b  = (__bf16*)(e2n + (size_t)Nn * 64);
  __bf16* X2b  = X1b + (size_t)rowsE * 128;
  __bf16* Y1b  = X2b + (size_t)rowsE * 64;
  __bf16* Y2b  = Y1b + (size_t)Nn * 128;

  float* out_node = (float*)d_out;
  float* out_edge = out_node + (size_t)Nn * 64;

  const int be = (rowsE + 63) / 64;
  const int bn = (Nn + 63) / 64;

  // zero stats + e2n (contiguous)
  k_zero<<<512, 256, 0, stream>>>(stats, 16LL * 128 + (long long)Nn * 64);

  // edge path
  k_edge_gemm1<<<be, 128, 0, stream>>>(edge_rep, node_rep, edge_nodes, eW1,
                                       X1b, stats + 0, stats + 128, rowsE);
  k_finalize<<<1, 128, 0, stream>>>(stats, 0, 128, 1.f / rowsE, eg1, eb1);
  k_gemm2<<<be, 128, 0, stream>>>(X1b, stats + 2 * 128, stats + 3 * 128, eW2,
                                  X2b, stats + 4 * 128, stats + 5 * 128, rowsE);
  k_finalize<<<1, 64, 0, stream>>>(stats, 4, 64, 1.f / rowsE, eg2, eb2);
  k_edge_out<<<2048, 256, 0, stream>>>(X2b, stats + 6 * 128, stats + 7 * 128,
                                       edge_nodes, out_edge, e2n,
                                       (long long)rowsE * 64);

  // node path
  k_node_gemm1<<<bn, 128, 0, stream>>>(node_rep, e2n, nW1, Y1b,
                                       stats + 8 * 128, stats + 9 * 128, Nn);
  k_finalize<<<1, 128, 0, stream>>>(stats, 8, 128, 1.f / Nn, ng1, nb1);
  k_gemm2<<<bn, 128, 0, stream>>>(Y1b, stats + 10 * 128, stats + 11 * 128, nW2,
                                  Y2b, stats + 12 * 128, stats + 13 * 128, Nn);
  k_finalize<<<1, 64, 0, stream>>>(stats, 12, 64, 1.f / Nn, ng2, nb2);
  k_node_out<<<1024, 256, 0, stream>>>(Y2b, stats + 14 * 128, stats + 15 * 128,
                                       out_node, (long long)Nn * 64);
}
